// MultiHeadAttention_54992761258357
// MI455X (gfx1250) — matrix operand order repaired and checked by reading, not compiled, not run on hardware
//
#include <hip/hip_runtime.h>
#include <hip/hip_bf16.h>

typedef __bf16 bf16_t;
typedef __bf16 v16bf __attribute__((ext_vector_type(16)));
typedef float  v8f   __attribute__((ext_vector_type(8)));

// Problem constants (match reference)
constexpr int Bc = 2, Sc = 2048, Dc = 1024, Hc = 16, HDc = 64;
constexpr int BS = Bc * Sc;                                // 4096 rows
constexpr size_t HEAD_ELEMS = (size_t)Bc * Hc * Sc * HDc;  // 4,194,304

// Q pre-scale: (1/sqrt(HD)) * log2(e) -> softmax runs in exp2 domain
#define QSCALE  0.18033688f            // 0.125 * 1.44269504
#define MASK_C2 (-9e-13f * 1.44269504f)

// ---------------------------------------------------------------------------
// Raw VALU helpers (values are never sNaN here; skip canonicalization).
// ---------------------------------------------------------------------------
__device__ __forceinline__ float vmax_raw(float a, float b) {
    float d;
    asm("v_max_num_f32_e32 %0, %1, %2" : "=v"(d) : "v"(a), "v"(b));
    return d;
}
__device__ __forceinline__ float fast_exp2(float x) {
    return __builtin_amdgcn_exp2f(x);      // v_exp_f32 (base-2), 1 TRANS op
}

// Fused DPP16 butterfly reductions over each 16-lane group: 4 VALU ops total.
__device__ __forceinline__ float dpp_max16(float v) {
    float t;
    asm("v_max_num_f32_dpp %0, %1, %1 quad_perm:[1,0,3,2] row_mask:0xf bank_mask:0xf bound_ctrl:1"
        : "=v"(t) : "v"(v));
    asm("v_max_num_f32_dpp %0, %1, %1 quad_perm:[2,3,0,1] row_mask:0xf bank_mask:0xf bound_ctrl:1"
        : "=v"(v) : "v"(t));
    asm("v_max_num_f32_dpp %0, %1, %1 row_half_mirror row_mask:0xf bank_mask:0xf bound_ctrl:1"
        : "=v"(t) : "v"(v));
    asm("v_max_num_f32_dpp %0, %1, %1 row_mirror row_mask:0xf bank_mask:0xf bound_ctrl:1"
        : "=v"(v) : "v"(t));
    return v;
}
__device__ __forceinline__ float dpp_sum16(float v) {
    float t;
    asm("v_add_f32_dpp %0, %1, %1 quad_perm:[1,0,3,2] row_mask:0xf bank_mask:0xf bound_ctrl:1"
        : "=v"(t) : "v"(v));
    asm("v_add_f32_dpp %0, %1, %1 quad_perm:[2,3,0,1] row_mask:0xf bank_mask:0xf bound_ctrl:1"
        : "=v"(v) : "v"(t));
    asm("v_add_f32_dpp %0, %1, %1 row_half_mirror row_mask:0xf bank_mask:0xf bound_ctrl:1"
        : "=v"(t) : "v"(v));
    asm("v_add_f32_dpp %0, %1, %1 row_mirror row_mask:0xf bank_mask:0xf bound_ctrl:1"
        : "=v"(v) : "v"(t));
    return v;
}

// ---------------------------------------------------------------------------
// Tiled bf16 WMMA GEMM: C[M,N] = A[M,K] * B[K,N] (+bias epilogue)
// Block: 256 threads = 8 waves; block tile 128x128; K-step 32.
// Register prefetch + double-buffered LDS, one barrier per K-step.
// ---------------------------------------------------------------------------
#define LDSA 40    // 32 + 8 pad (bf16 elems)
#define LDSB 136   // 128 + 8 pad

template<bool A_F32, bool QKV_EPI>
__global__ __launch_bounds__(256)
void gemm_wmma_kernel(const void* __restrict__ Ap,
                      const float* __restrict__ Bp,
                      const float* __restrict__ bias,
                      float* __restrict__ Cout,
                      bf16_t* __restrict__ Qo, bf16_t* __restrict__ Ko,
                      bf16_t* __restrict__ Vto,
                      int M, int N, int K)
{
    __shared__ bf16_t As[2][128 * LDSA];
    __shared__ bf16_t Bs[2][32 * LDSB];

    const int tid  = threadIdx.x;
    const int lane = tid & 31;
    const int wave = tid >> 5;           // 0..7
    const int wm   = wave >> 2;          // 0..1 (row group, 64 rows each)
    const int wn   = wave & 3;           // 0..3 (col group, 32 cols each)
    const int l16  = lane & 15;
    const int lhi  = lane >> 4;

    const int bm = blockIdx.y * 128;
    const int bn = blockIdx.x * 128;

    v8f acc[4][2];
#pragma unroll
    for (int mt = 0; mt < 4; ++mt)
#pragma unroll
        for (int nt = 0; nt < 2; ++nt)
#pragma unroll
            for (int r = 0; r < 8; ++r) acc[mt][nt][r] = 0.0f;

    // global->LDS staging indices
    const int ar = tid >> 1;            // 0..127 (A row)
    const int ac = (tid & 1) * 16;      // 16 consecutive K elems
    const int br = tid >> 3;            // 0..31  (B row = k)
    const int bc = (tid & 7) * 16;      // 16 consecutive N elems

    float  aregf[16];
    bf16_t aregb[16];
    float  bregf[16];

    auto load_regs = [&](int k0) {
        if constexpr (A_F32) {
            const float* src = (const float*)Ap + (size_t)(bm + ar) * K + k0 + ac;
#pragma unroll
            for (int i = 0; i < 16; ++i) aregf[i] = src[i];
        } else {
            const bf16_t* src = (const bf16_t*)Ap + (size_t)(bm + ar) * K + k0 + ac;
#pragma unroll
            for (int i = 0; i < 16; ++i) aregb[i] = src[i];
        }
        const float* bsrc = Bp + (size_t)(k0 + br) * N + bn + bc;
#pragma unroll
        for (int i = 0; i < 16; ++i) bregf[i] = bsrc[i];
    };
    auto store_lds = [&](int buf) {
#pragma unroll
        for (int i = 0; i < 16; ++i) {
            if constexpr (A_F32) As[buf][ar * LDSA + ac + i] = (bf16_t)aregf[i];
            else                 As[buf][ar * LDSA + ac + i] = aregb[i];
        }
#pragma unroll
        for (int i = 0; i < 16; ++i)
            Bs[buf][br * LDSB + bc + i] = (bf16_t)bregf[i];
    };

    load_regs(0);
    store_lds(0);
    __syncthreads();

    int buf = 0;
    for (int k0 = 0; k0 < K; k0 += 32) {
        const bool has_next = (k0 + 32 < K);
        if (has_next) load_regs(k0 + 32);          // prefetch next tile to regs

        // ---- build fragments (contiguous k = 16*lhi + i in A and B, wave32) ----
        v16bf afrag[4], bfrag[2];
#pragma unroll
        for (int mt = 0; mt < 4; ++mt) {
            const int row  = wm * 64 + mt * 16 + l16;
            const bf16_t* p = &As[buf][row * LDSA + lhi * 16];
#pragma unroll
            for (int i = 0; i < 8; ++i) {
                afrag[mt][i]     = p[i];        // K = koff + i
                afrag[mt][8 + i] = p[8 + i];    // K = koff + 8 + i
            }
        }
#pragma unroll
        for (int nt = 0; nt < 2; ++nt) {
            const int col = wn * 32 + nt * 16 + l16;
            const int kb  = lhi * 16;
#pragma unroll
            for (int i = 0; i < 16; ++i)
                bfrag[nt][i] = Bs[buf][(kb + i) * LDSB + col];
        }

#pragma unroll
        for (int mt = 0; mt < 4; ++mt)
#pragma unroll
            for (int nt = 0; nt < 2; ++nt)
                acc[mt][nt] = __builtin_amdgcn_wmma_f32_16x16x32_bf16(
                    false, afrag[mt], false, bfrag[nt],
                    (short)0, acc[mt][nt], false, false);

        if (has_next) {
            store_lds(buf ^ 1);                    // fill other buffer
            __syncthreads();
            buf ^= 1;
        }
    }

    // ---- epilogue (branchless for QKV scatter) ----
#pragma unroll
    for (int mt = 0; mt < 4; ++mt) {
#pragma unroll
        for (int nt = 0; nt < 2; ++nt) {
            const int ncol = bn + wn * 32 + nt * 16 + l16;
            const float bv = bias[ncol];
#pragma unroll
            for (int r = 0; r < 8; ++r) {
                const int m = bm + wm * 64 + mt * 16 + r + lhi * 8;
                const float v = acc[mt][nt][r] + bv;
                if (QKV_EPI) {
                    // n in [0,3072): [H][3*HD] -> Q/K [B,H,S,HD], V transposed [B,H,HD,S]
                    const int b   = m >> 11;         // S = 2048
                    const int s   = m & 2047;
                    const int h   = ncol / 192;
                    const int rr  = ncol - h * 192;
                    const int sel = rr >> 6;         // 0=q 1=k 2=v
                    const int d   = rr & 63;
                    const size_t bhh    = (size_t)(b * Hc + h);
                    const size_t idx_qk = (bhh * Sc + s) * HDc + d;
                    const size_t idx_v  = (bhh * HDc + d) * Sc + s;
                    bf16_t* base = (sel == 2) ? Vto : ((sel == 0) ? Qo : Ko);
                    const size_t idx = (sel == 2) ? idx_v : idx_qk;
                    const float  vs  = (sel == 0) ? v * QSCALE : v;
                    base[idx] = (bf16_t)vs;          // single store, no branches
                } else {
                    Cout[(size_t)m * N + ncol] = v;
                }
            }
        }
    }
}

// ---------------------------------------------------------------------------
// Suffix sums of V rows at 32-key block granularity:
// Suf[bh][j][d] = sum_{k >= 32*j} V[bh][k][d]   (f32)
// ---------------------------------------------------------------------------
__global__ __launch_bounds__(256)
void vsuffix_kernel(const bf16_t* __restrict__ Vt, float* __restrict__ Suf)
{
    const int idx = blockIdx.x * 256 + threadIdx.x;   // 0..2047
    const int bh = idx >> 6;
    const int d  = idx & 63;
    const bf16_t* row = Vt + ((size_t)bh * HDc + d) * Sc;
    float acc = 0.0f;
    for (int k = Sc - 1; k >= 0; --k) {
        acc += (float)row[k];
        if ((k & 31) == 0)
            Suf[((size_t)bh * (Sc / 32) + (k >> 5)) * HDc + d] = acc;
    }
}

// ---------------------------------------------------------------------------
// Flash-style attention, exp2-domain softmax, causal-split:
//  - blocks fully below diagonal: no mask code
//  - one straddling block per q-tile: element mask
//  - fully-masked tail (uniform log2-logit MASK_C2): analytic via V suffix sums
// Q(prescaled),K: bf16 [B*H][S][64], Vt: bf16 [B*H][64][S], O: bf16 [B][S][H*64]
// ---------------------------------------------------------------------------
#define PSTR 34   // 32 + 2 pad

__global__ __launch_bounds__(128)
void attn_kernel(const bf16_t* __restrict__ Q, const bf16_t* __restrict__ K,
                 const bf16_t* __restrict__ Vt, const float* __restrict__ Suf,
                 bf16_t* __restrict__ O)
{
    __shared__ bf16_t Pl[4 * 16 * PSTR];   // per-wave P staging

    const int lane = threadIdx.x & 31;
    const int wave = threadIdx.x >> 5;     // 0..3
    const int l16  = lane & 15;
    const int lhi  = lane >> 4;

    const int bh = blockIdx.y;             // 0..31
    const int b  = bh >> 4;
    const int h  = bh & 15;
    // wave-uniform by construction; force scalar for SALU loop control
    const int qbase = __builtin_amdgcn_readfirstlane(blockIdx.x * 64 + wave * 16);

    const bf16_t* Qh  = Q  + (size_t)bh * Sc * HDc;
    const bf16_t* Kh  = K  + (size_t)bh * Sc * HDc;
    const bf16_t* Vth = Vt + (size_t)bh * HDc * Sc;
    bf16_t* Pw = &Pl[(threadIdx.x >> 5) * 16 * PSTR];

    // ---- Q A-fragments (HD=64 -> 2 K-chunks of 32) ----
    v16bf qf[2];
    {
        const int qrow = qbase + l16;
#pragma unroll
        for (int c = 0; c < 2; ++c) {
            const bf16_t* p = Qh + (size_t)qrow * HDc + c * 32 + lhi * 16;
#pragma unroll
            for (int i = 0; i < 8; ++i) {
                qf[c][i]     = p[i];
                qf[c][8 + i] = p[8 + i];
            }
        }
    }

    float m_[8], l_[8];
    v8f o[4];
#pragma unroll
    for (int r = 0; r < 8; ++r) { m_[r] = -INFINITY; l_[r] = 0.0f; }
#pragma unroll
    for (int nc = 0; nc < 4; ++nc)
#pragma unroll
        for (int r = 0; r < 8; ++r) o[nc][r] = 0.0f;

    auto block_step = [&](const int kb, const bool masked) {
        // ---- scores (log2 domain; scale pre-folded into Q) ----
        v8f sc[2];
#pragma unroll
        for (int nt = 0; nt < 2; ++nt) {
            v8f z;
#pragma unroll
            for (int r = 0; r < 8; ++r) z[r] = 0.0f;
#pragma unroll
            for (int c = 0; c < 2; ++c) {
                const int key = kb + nt * 16 + l16;
                v16bf kf = *(const v16bf*)(Kh + (size_t)key * HDc + c * 32 + lhi * 16);
                z = __builtin_amdgcn_wmma_f32_16x16x32_bf16(
                        false, qf[c], false, kf, (short)0, z, false, false);
            }
            sc[nt] = z;
        }

        // ---- (+mask) + online softmax update, exp2 domain ----
#pragma unroll
        for (int r = 0; r < 8; ++r) {
            float v0 = sc[0][r];
            float v1 = sc[1][r];
            if (masked) {
                const int qi = qbase + r + lhi * 8;
                if (kb + l16 > qi)      v0 = MASK_C2;
                if (kb + 16 + l16 > qi) v1 = MASK_C2;
            }

            const float t  = dpp_max16(vmax_raw(v0, v1));
            const float mn = vmax_raw(m_[r], t);
            const float alpha = fast_exp2(m_[r] - mn);
            const float p0 = fast_exp2(v0 - mn);
            const float p1 = fast_exp2(v1 - mn);
            const float rs = dpp_sum16(p0 + p1);

            l_[r] = l_[r] * alpha + rs;
            m_[r] = mn;
#pragma unroll
            for (int nc = 0; nc < 4; ++nc) o[nc][r] *= alpha;

            const int row = r + lhi * 8;
            Pw[row * PSTR + l16]      = (bf16_t)p0;
            Pw[row * PSTR + 16 + l16] = (bf16_t)p1;
        }

        // LDS stores -> fragment loads are same-wave in-order; wait explicitly
        asm volatile("s_wait_dscnt 0x0" ::: "memory");

        // ---- P as A-fragment (16x32) ----
        v16bf pf;
        {
            const bf16_t* pp = Pw + l16 * PSTR + lhi * 16;
#pragma unroll
            for (int i = 0; i < 8; ++i) {
                pf[i]     = pp[i];
                pf[8 + i] = pp[8 + i];
            }
        }

        // ---- PV: 4 WMMAs over HD=64; Vt gives contiguous B-fragments ----
#pragma unroll
        for (int nc = 0; nc < 4; ++nc) {
            v16bf vf = *(const v16bf*)(Vth + (size_t)(nc * 16 + l16) * Sc + kb + lhi * 16);
            o[nc] = __builtin_amdgcn_wmma_f32_16x16x32_bf16(
                        false, pf, false, vf, (short)0, o[nc], false, false);
        }
    };

    // last block index that contains any key <= qbase+15 (scalar)
    const int kb_last = __builtin_amdgcn_readfirstlane(((qbase + 15) >> 5) << 5);

    for (int kb = 0; kb < kb_last; kb += 32) block_step(kb, false);  // unmasked
    block_step(kb_last, true);                                       // diagonal

    // ---- analytic fully-masked tail: uniform log2-logit MASK_C2 ----
    const int kend = kb_last + 32;
    const int nrem = Sc - kend;
    if (nrem > 0) {
        float sv[4];
#pragma unroll
        for (int nc = 0; nc < 4; ++nc)
            sv[nc] = Suf[((size_t)bh * (Sc / 32) + (kend >> 5)) * HDc + nc * 16 + l16];
        const float fn = (float)nrem;
#pragma unroll
        for (int r = 0; r < 8; ++r) {
            const float mn = vmax_raw(m_[r], MASK_C2);
            const float alpha = fast_exp2(m_[r] - mn);
            const float pt = fast_exp2(MASK_C2 - mn);
            l_[r] = l_[r] * alpha + fn * pt;
            m_[r] = mn;
#pragma unroll
            for (int nc = 0; nc < 4; ++nc)
                o[nc][r] = o[nc][r] * alpha + pt * sv[nc];
        }
    }

    // ---- normalize + store to [B,S,D] bf16 ----
#pragma unroll
    for (int r = 0; r < 8; ++r) {
        const float inv = 1.0f / l_[r];
        const int qi = qbase + r + lhi * 8;
#pragma unroll
        for (int nc = 0; nc < 4; ++nc) {
            const int d = h * HDc + nc * 16 + l16;
            O[((size_t)b * Sc + qi) * Dc + d] = (bf16_t)(o[nc][r] * inv);
        }
    }
}

// ---------------------------------------------------------------------------
extern "C" void kernel_launch(void* const* d_in, const int* in_sizes, int n_in,
                              void* d_out, int out_size, void* d_ws, size_t ws_size,
                              hipStream_t stream) {
    const float* x     = (const float*)d_in[0];
    const float* W_qkv = (const float*)d_in[1];
    const float* b_qkv = (const float*)d_in[2];
    const float* W_out = (const float*)d_in[3];
    const float* b_out = (const float*)d_in[4];
    float* out = (float*)d_out;

    bf16_t* ws = (bf16_t*)d_ws;
    bf16_t* Qb  = ws;
    bf16_t* Kb  = ws + HEAD_ELEMS;
    bf16_t* Vtb = ws + 2 * HEAD_ELEMS;          // transposed V [B,H,HD,S]
    bf16_t* Ob  = ws + 3 * HEAD_ELEMS;          // attention out bf16 [B,S,D]
    float*  Suf = (float*)(ws + 4 * HEAD_ELEMS);// [B*H][S/32][HD] f32 (512 KB)

    // 1) QKV projection + bias + head-split scatter (V transposed, Q prescaled)
    gemm_wmma_kernel<true, true><<<dim3(3 * Dc / 128, BS / 128), 256, 0, stream>>>(
        x, W_qkv, b_qkv, nullptr, Qb, Kb, Vtb, BS, 3 * Dc, Dc);

    // 2) V suffix sums (for analytic masked tail)
    vsuffix_kernel<<<dim3(Bc * Hc * HDc / 256), 256, 0, stream>>>(Vtb, Suf);

    // 3) attention (flash-style, exp2-domain softmax, causal-split)
    attn_kernel<<<dim3(Sc / 64, Bc * Hc), 128, 0, stream>>>(Qb, Kb, Vtb, Suf, Ob);

    // 4) output projection + bias (f32 out)
    gemm_wmma_kernel<false, false><<<dim3(Dc / 128, BS / 128), 256, 0, stream>>>(
        Ob, W_out, b_out, out, nullptr, nullptr, nullptr, BS, Dc, Dc);
}
